// RelationalMemoryNeuro_90314572300603
// MI455X (gfx1250) — matrix operand order repaired and checked by reading, not compiled, not run on hardware
//
#include <hip/hip_runtime.h>
#include <stdint.h>

// ---------------- problem constants (from reference) ----------------
#define NCON   4096      // N_CONCEPTS
#define NREL   8
#define RANK   16
#define DDIM   1024      // feature dim, sqrt(D) = 32
#define TOPK   128
#define ROWS   8192      // B*T = 4*2048
#define MTILE  16        // rows per workgroup (one WMMA M tile)
#define NPAD   (NCON + 2)   // padded sim stride (bank-conflict avoidance)
#define XPAD   (DDIM + 8)   // padded x-tile stride (bank spread for ds reads)

typedef __attribute__((ext_vector_type(16))) __bf16 v16bf;
typedef __attribute__((ext_vector_type(8)))  float  v8f;

// float -> bf16 (round to nearest even), self-contained
__device__ __forceinline__ unsigned f32_to_bf16_bits(float f) {
    union { float f; uint32_t u; } c; c.f = f;
    uint32_t u = c.u;
    u += 0x7fffu + ((u >> 16) & 1u);
    return u >> 16;
}

// monotonic key: descending float order == descending unsigned order
__device__ __forceinline__ unsigned key_of(float f) {
    union { float f; unsigned u; } c; c.f = f;
    unsigned u = c.u;
    return (u & 0x80000000u) ? ~u : (u | 0x80000000u);
}

// ---------------- kernel 1: proto fp32 -> bf16 staging (x converts in-kernel) ----------------
__global__ void cvt_proto_kernel(const float* __restrict__ proto,
                                 unsigned short* __restrict__ ph) {
    size_t i = (size_t)blockIdx.x * blockDim.x + threadIdx.x;
    if (i < (size_t)NCON * DDIM)
        ph[i] = (unsigned short)f32_to_bf16_bits(proto[i]);
}

// ---------------- kernel 2: fused sim-GEMM + topk + softmax + low-rank + ctx ----------------
__global__ __launch_bounds__(256, 1)
void relmem_fused_kernel(const unsigned short* __restrict__ ph,
                         const float* __restrict__ x,
                         const float* __restrict__ proto,
                         const float* __restrict__ A,
                         const float* __restrict__ Bm,
                         const float* __restrict__ gains,
                         float* __restrict__ y) {
    __shared__ float          simS[MTILE][NPAD];   // 256.1 KB: full 16x4096 sim tile
    __shared__ unsigned short xLds[MTILE][XPAD];   // 33.0 KB: bf16 X tile (A operand)
    __shared__ unsigned histS[256];
    __shared__ unsigned escS[256];
    __shared__ float    redS[256];
    __shared__ int      topIdxS[TOPK];
    __shared__ float    topValS[TOPK];
    __shared__ float    attS[TOPK];
    __shared__ float    wS[TOPK];
    __shared__ float    gS[NREL];
    __shared__ float    denomS;
    __shared__ unsigned prefixS;
    __shared__ int      remS;

    const int tid    = threadIdx.x;
    const int wave   = tid >> 5;
    const int lane   = tid & 31;
    const int lane16 = lane & 15;
    const int hi     = lane >> 4;
    const int mBase  = blockIdx.x * MTILE;

    if (tid < NREL) gS[tid] = gains[tid];
    if (tid == 0) {
        float s = 0.f;
        #pragma unroll
        for (int l = 0; l < NREL; ++l) s += gains[l];
        denomS = (s > 0.f) ? s : 1.f;
    }

    // ---- stage X tile into LDS, converting fp32 -> bf16 on the fly ----
    for (int i = tid; i < MTILE * (DDIM / 4); i += 256) {
        const int row = i >> 8;            // / (DDIM/4)
        const int col = (i & 255) * 4;
        const float4 f = *(const float4*)(x + (size_t)(mBase + row) * DDIM + col);
        uint2 p;
        p.x = f32_to_bf16_bits(f.x) | (f32_to_bf16_bits(f.y) << 16);
        p.y = f32_to_bf16_bits(f.z) | (f32_to_bf16_bits(f.w) << 16);
        *(uint2*)&xLds[row][col] = p;
    }
    __syncthreads();

    // ============ Phase 1: sim[16][4096] = (Xtile . protoT) / 32 via WMMA bf16 ============
    // A fragment (16x32 bf16): lane<16 holds K {0..7,16..23}, lane>=16 holds K {8..15,24..31}
    // B fragment (32x16 bf16): lane n holds proto row (nBase+n), same K split as A
    const float simScale = 1.0f / 32.0f;   // 1/sqrt(D)
    for (int nt = wave; nt < NCON / 16; nt += 8) {
        const int nBase = nt * 16;
        const unsigned short* bRow = ph + (size_t)(nBase + lane16) * DDIM;
        v8f acc = {};
        for (int k0 = 0; k0 < DDIM; k0 += 32) {
            union { v16bf v; uint4 q[2]; } af, bf;
            af.q[0] = *(const uint4*)&xLds[lane16][k0 + hi * 8];
            af.q[1] = *(const uint4*)&xLds[lane16][k0 + 16 + hi * 8];
            bf.q[0] = *(const uint4*)(bRow + k0 + hi * 8);
            bf.q[1] = *(const uint4*)(bRow + k0 + 16 + hi * 8);
            acc = __builtin_amdgcn_wmma_f32_16x16x32_bf16(
                false, af.v, false, bf.v, (short)0, acc, false, false);
        }
        // C/D layout: vgpr r -> (M = r + 8*hi, N = lane16)
        #pragma unroll
        for (int r = 0; r < 8; ++r)
            simS[r + 8 * hi][nBase + lane16] = acc[r] * simScale;
    }
    __syncthreads();

    // ============ Phase 2..5 per row ============
    for (int m = 0; m < MTILE; ++m) {
        const float* srow = simS[m];

        // ---- exact top-128 threshold via 4-level radix select on monotonic keys ----
        if (tid == 0) { prefixS = 0u; remS = TOPK; }
        for (int lv = 3; lv >= 0; --lv) {
            histS[tid] = 0u;
            __syncthreads();
            const int sh = lv * 8;
            const unsigned mask = (lv == 3) ? 0u : (0xFFFFFFFFu << (sh + 8));
            const unsigned pfx = prefixS;
            for (int i = tid; i < NCON; i += 256) {
                unsigned key = key_of(srow[i]);
                if (((key ^ pfx) & mask) == 0u)
                    atomicAdd(&histS[(key >> sh) & 255u], 1u);  // int adds: order-free
            }
            __syncthreads();
            if (tid == 0) {
                int rem = remS;
                int b = 255;
                for (; b > 0; --b) {
                    int c = (int)histS[b];
                    if (c >= rem) break;
                    rem -= c;
                }
                prefixS |= (unsigned)b << sh;
                remS = rem;
            }
            __syncthreads();
        }
        const unsigned thr = prefixS;   // key of the 128th-largest element
        const int nties = remS;         // how many ==thr to accept
        const int G = TOPK - nties;     // strictly-greater count

        // ---- deterministic collection: count -> exclusive scan -> index-ordered place ----
        const int ibase = tid * (NCON / 256);   // 16 contiguous elements per thread
        {
            int myG = 0, myE = 0;
            #pragma unroll
            for (int j = 0; j < NCON / 256; ++j) {
                unsigned key = key_of(srow[ibase + j]);
                if (key > thr) ++myG;
                else if (key == thr) ++myE;
            }
            histS[tid] = (unsigned)myG;
            escS[tid]  = (unsigned)myE;
        }
        __syncthreads();
        if (tid == 0) {   // serial exclusive scan over 256 counters
            unsigned aG = 0, aE = 0;
            for (int t = 0; t < 256; ++t) {
                unsigned g = histS[t]; histS[t] = aG; aG += g;
                unsigned e = escS[t];  escS[t]  = aE; aE += e;
            }
        }
        __syncthreads();
        {
            int gpos = (int)histS[tid];
            int epos = (int)escS[tid];
            #pragma unroll
            for (int j = 0; j < NCON / 256; ++j) {
                const int i = ibase + j;
                const float v = srow[i];
                const unsigned key = key_of(v);
                if (key > thr) {
                    topIdxS[gpos] = i; topValS[gpos] = v; ++gpos;
                } else if (key == thr) {
                    if (epos < nties) { topIdxS[G + epos] = i; topValS[G + epos] = v; }
                    ++epos;
                }
            }
        }
        __syncthreads();

        // ---- softmax over the 128 selected logits (fixed-tree reductions) ----
        float v = (tid < TOPK) ? topValS[tid] : -3.402823466e38f;
        redS[tid] = v;
        __syncthreads();
        #pragma unroll
        for (int s = 128; s > 0; s >>= 1) {
            if (tid < s) redS[tid] = fmaxf(redS[tid], redS[tid + s]);
            __syncthreads();
        }
        const float mx = redS[0];
        __syncthreads();
        float e = (tid < TOPK) ? __expf(v - mx) : 0.f;
        redS[tid] = e;
        __syncthreads();
        #pragma unroll
        for (int s = 128; s > 0; s >>= 1) {
            if (tid < s) redS[tid] += redS[tid + s];
            __syncthreads();
        }
        const float invSum = 1.0f / redS[0];
        __syncthreads();
        if (tid < TOPK) attS[tid] = e * invSum;
        if (tid < TOPK) wS[tid] = 0.f;
        __syncthreads();

        // ---- low-rank relational term: z = att.A_sel ; c = z.B_selT ; w += g*c ----
        const int r  = tid & 15;    // rank lane
        const int kk = tid >> 4;    // 16 k-slices
        for (int l = 0; l < NREL; ++l) {
            const size_t base = (size_t)l * NCON * RANK;
            float part = 0.f;
            #pragma unroll
            for (int k = 0; k < TOPK / 16; ++k) {
                const int ks = kk + k * 16;
                part += attS[ks] * A[base + (size_t)topIdxS[ks] * RANK + r];
            }
            redS[tid] = part;                 // redS[kk*16 + r]
            __syncthreads();
            #pragma unroll
            for (int s = 8; s > 0; s >>= 1) { // reduce over kk: fixed tree
                if (kk < s) redS[tid] += redS[tid + s * 16];
                __syncthreads();
            }
            // redS[0..15] now hold z[r]
            if (tid < TOPK) {
                const float* Br = Bm + ((size_t)l * NCON + topIdxS[tid]) * RANK;
                float c = 0.f;
                #pragma unroll
                for (int r2 = 0; r2 < RANK; ++r2) c += redS[r2] * Br[r2];
                wS[tid] += gS[l] * c;
            }
            __syncthreads();
        }

        // ---- ctx gather-matvec + residual (float4 lanes): ----
        // y = x + (0.1/denom) * sum_k w[k]*proto[idx[k]]
        const int row = mBase + m;
        const float coef = 0.1f / denomS;
        const int d0 = tid * 4;               // 256 threads * 4 = 1024 dims
        float4 acc = make_float4(0.f, 0.f, 0.f, 0.f);
        for (int k = 0; k < TOPK; ++k) {
            const float wk = wS[k];
            const float4 p = *(const float4*)(proto + (size_t)topIdxS[k] * DDIM + d0);
            acc.x += wk * p.x; acc.y += wk * p.y;
            acc.z += wk * p.z; acc.w += wk * p.w;
        }
        const float4 xv = *(const float4*)(x + (size_t)row * DDIM + d0);
        float4 out;
        out.x = xv.x + coef * acc.x;
        out.y = xv.y + coef * acc.y;
        out.z = xv.z + coef * acc.z;
        out.w = xv.w + coef * acc.w;
        *(float4*)(y + (size_t)row * DDIM + d0) = out;
        __syncthreads();
    }
}

// ---------------- host-side launch ----------------
extern "C" void kernel_launch(void* const* d_in, const int* in_sizes, int n_in,
                              void* d_out, int out_size, void* d_ws, size_t ws_size,
                              hipStream_t stream) {
    (void)in_sizes; (void)n_in; (void)out_size; (void)ws_size;
    const float* x     = (const float*)d_in[0];
    const float* proto = (const float*)d_in[1];
    const float* A     = (const float*)d_in[2];
    const float* Bm    = (const float*)d_in[3];
    const float* gains = (const float*)d_in[4];
    // d_in[5] is top_k == 128 (compile-time constant here)
    float* y = (float*)d_out;

    unsigned short* ph = (unsigned short*)d_ws;       // 4096*1024 bf16 = 8.4 MB

    const size_t NP = (size_t)NCON * DDIM;
    cvt_proto_kernel<<<(int)((NP + 255) / 256), 256, 0, stream>>>(proto, ph);
    relmem_fused_kernel<<<ROWS / MTILE, 256, 0, stream>>>(ph, x, proto, A, Bm, gains, y);
}
